// TopoClassifier_62474594287651
// MI455X (gfx1250) — compile-verified, matrix operand order checked by reading, not run on hardware
//
#include <hip/hip_runtime.h>
#include <cfloat>
#include <cstdint>
#include <cmath>

// Problem sizes (fixed by reference setup_inputs)
#define B_    32
#define C_    64
#define H_    256
#define W_    256
#define HW_   (H_ * W_)
#define HID_  32
#define MLP_  128
#define NC_   5
#define EPSW  1e-6f
#define EPSP  1e-12f

typedef __attribute__((ext_vector_type(16))) _Float16 v16h;
typedef __attribute__((ext_vector_type(8)))  float    v8f;
typedef __attribute__((ext_vector_type(4)))  int      v4i;

// ---------------------------------------------------------------------------
// WMMA f16 16x16x32 A/B element -> K mapping (per CDNA5 ISA 7.12.2):
// lanes 0-15 hold K {0..7,16..23}, lanes 16-31 hold K {8..15,24..31}.
// ---------------------------------------------------------------------------
__device__ __forceinline__ int kmap(int lane, int e) {
  int lo = (lane < 16) ? 0 : 8;
  return (e < 8) ? (lo + e) : (16 + lo + (e - 8));
}

// ---------------------------------------------------------------------------
// Kernel 1: z[b,c] = mean over HxW  (one pass over x)
// ---------------------------------------------------------------------------
__global__ void k_mean(const float* __restrict__ x, float* __restrict__ z) {
  const int slice = blockIdx.x;                    // b*C + c
  const float4* g4 = (const float4*)(x + (size_t)slice * HW_);
  float s = 0.f;
  for (int i = threadIdx.x; i < HW_ / 4; i += blockDim.x) {
    float4 v = g4[i];
    s += v.x + v.y + v.z + v.w;
  }
  __shared__ float red[32];
  for (int off = 16; off > 0; off >>= 1) s += __shfl_down(s, off, 32);
  const int wave = threadIdx.x >> 5, lane = threadIdx.x & 31;
  if (lane == 0) red[wave] = s;
  __syncthreads();
  if (wave == 0) {
    float t = (lane < (int)(blockDim.x >> 5)) ? red[lane] : 0.f;
    for (int off = 16; off > 0; off >>= 1) t += __shfl_down(t, off, 32);
    if (lane == 0) z[slice] = t / (float)HW_;
  }
}

// ---------------------------------------------------------------------------
// Kernel 2: channel attention MLP + softmax, single wave, WMMA f16
//   h = relu(z @ fc1_w.T + b1)   [32,32]
//   s = h @ fc2_w.T + b2         [32,64]
//   w = softmax(s, axis=1)
// ---------------------------------------------------------------------------
__global__ void k_attn(const float* __restrict__ z,
                       const float* __restrict__ fc1_w, const float* __restrict__ fc1_b,
                       const float* __restrict__ fc2_w, const float* __restrict__ fc2_b,
                       float* __restrict__ w_ws, float* __restrict__ w_out) {
  __shared__ float zs[B_][C_];
  __shared__ float hs[B_][HID_];
  __shared__ float ss[B_][C_];
  const int lane = threadIdx.x;          // 0..31 (one wave, EXEC all ones)
  for (int c = 0; c < C_; ++c) zs[lane][c] = z[lane * C_ + c];
  __syncthreads();

  const int m15  = lane & 15;
  const int hsel = (lane < 16) ? 0 : 8;

  // GEMM1: [32,64] x [64,32] -> [32,32]
  for (int mt = 0; mt < 2; ++mt)
    for (int nt = 0; nt < 2; ++nt) {
      v8f acc = {};
      for (int kc = 0; kc < 2; ++kc) {
        v16h a, b;
        for (int e = 0; e < 16; ++e) {
          int k = kc * 32 + kmap(lane, e);
          a[e] = (_Float16)zs[mt * 16 + m15][k];
          b[e] = (_Float16)fc1_w[(nt * 16 + m15) * C_ + k];   // B[k][n] = fc1_w[n][k]
        }
        acc = __builtin_amdgcn_wmma_f32_16x16x32_f16(false, a, false, b,
                                                     (short)0, acc, false, false);
      }
      for (int v = 0; v < 8; ++v) {
        int m = mt * 16 + hsel + v, n = nt * 16 + m15;
        float t = acc[v] + fc1_b[n];
        hs[m][n] = t > 0.f ? t : 0.f;
      }
    }
  __syncthreads();

  // GEMM2: [32,32] x [32,64] -> [32,64]
  for (int mt = 0; mt < 2; ++mt)
    for (int nt = 0; nt < 4; ++nt) {
      v8f acc = {};
      v16h a, b;
      for (int e = 0; e < 16; ++e) {
        int k = kmap(lane, e);
        a[e] = (_Float16)hs[mt * 16 + m15][k];
        b[e] = (_Float16)fc2_w[(nt * 16 + m15) * HID_ + k];   // B[k][n] = fc2_w[n][k]
      }
      acc = __builtin_amdgcn_wmma_f32_16x16x32_f16(false, a, false, b,
                                                   (short)0, acc, false, false);
      for (int v = 0; v < 8; ++v) {
        int m = mt * 16 + hsel + v, n = nt * 16 + m15;
        ss[m][n] = acc[v] + fc2_b[n];
      }
    }
  __syncthreads();

  // Row softmax: lane handles batch row `lane`
  float mx = -FLT_MAX;
  for (int c = 0; c < C_; ++c) mx = fmaxf(mx, ss[lane][c]);
  float sum = 0.f;
  for (int c = 0; c < C_; ++c) { float e = __expf(ss[lane][c] - mx); ss[lane][c] = e; sum += e; }
  for (int c = 0; c < C_; ++c) {
    float wv = ss[lane][c] / sum;
    w_ws [lane * C_ + c] = wv;
    w_out[lane * C_ + c] = wv;
  }
}

// ---------------------------------------------------------------------------
// Kernel 3: per-(B,C) topo entropy. Whole 256KB slice staged in LDS (CDNA5:
// 320KB/WGP) via async-to-LDS. Thread owns one column, rolls the 3x3 window
// over rows: lane-consecutive columns -> consecutive LDS banks (conflict-free).
// Since w>0 per channel: l = w * relu(pool(x)-x); w folded analytically.
// ---------------------------------------------------------------------------
#if __has_builtin(__builtin_amdgcn_global_load_async_to_lds_b128) && \
    __has_builtin(__builtin_amdgcn_s_wait_asynccnt)
#define TOPO_ASYNC_LDS 1
// Builtin signature (from compiler diagnostic): (v4i AS1*, v4i AS3*, imm, imm)
typedef __attribute__((address_space(1))) v4i* gptr128;
typedef __attribute__((address_space(3))) v4i* lptr128;
#endif

__device__ __forceinline__ float block_reduce_1024(float v, float* red, int tid) {
  for (int off = 16; off > 0; off >>= 1) v += __shfl_down(v, off, 32);
  const int wave = tid >> 5, lane = tid & 31;
  __syncthreads();                       // protect red[] reuse
  if (lane == 0) red[wave] = v;
  __syncthreads();
  if (wave == 0) {
    float t = red[lane];
    for (int off = 16; off > 0; off >>= 1) t += __shfl_down(t, off, 32);
    if (lane == 0) red[0] = t;
  }
  __syncthreads();
  return red[0];
}

__global__ __launch_bounds__(1024, 1)
void k_topo(const float* __restrict__ x, const float* __restrict__ w_ws,
            float* __restrict__ feats) {
  extern __shared__ float tile[];        // HW_ floats = 256 KB
  __shared__ float red[32];
  const int slice = blockIdx.x;
  const int tid   = threadIdx.x;
  const float* gx = x + (size_t)slice * HW_;

#ifdef TOPO_ASYNC_LDS
  for (int i = tid; i < HW_ / 4; i += 1024) {
    // Generic LDS pointer carries the LDS offset in its low 32 bits
    // (FLAT aperture rule); integer casts keep it.
    __builtin_amdgcn_global_load_async_to_lds_b128(
        (gptr128)(uintptr_t)(gx + 4 * (size_t)i),
        (lptr128)(uint32_t)(uintptr_t)(tile + 4 * (size_t)i),
        0, 0);
  }
  __builtin_amdgcn_s_wait_asynccnt(0);
#else
  {
    const float4* g4 = (const float4*)gx;
    float4* t4 = (float4*)tile;
    for (int i = tid; i < HW_ / 4; i += 1024) t4[i] = g4[i];
  }
#endif
  __syncthreads();

  const float wv = w_ws[slice];
  const int c   = tid & (W_ - 1);        // column 0..255 (lane-consecutive)
  const int r0  = (tid >> 8) * 64;       // 4 row-segments of 64 rows

  // horizontal 3-tap max/min at row r, columns c-1..c+1; also returns center
  auto h3 = [&](int r, float& hmx, float& hmn, float& ctr) {
    const float* rowp = tile + r * W_;
    float v = rowp[c];
    float mx = v, mn = v;
    if (c > 0)      { float a = rowp[c - 1]; mx = fmaxf(mx, a); mn = fminf(mn, a); }
    if (c < W_ - 1) { float a = rowp[c + 1]; mx = fmaxf(mx, a); mn = fminf(mn, a); }
    hmx = mx; hmn = mn; ctr = v;
  };

  // ---- pass 1: raw sums R0 = sum relu(mx-x), R1 = sum relu(x-mn) ----
  float R0 = 0.f, R1 = 0.f;
  {
    float hmxm, hmnm, hmx0, hmn0, hmxp, hmnp, xv0, xvp, dummy;
    if (r0 > 0) h3(r0 - 1, hmxm, hmnm, dummy); else { hmxm = -FLT_MAX; hmnm = FLT_MAX; }
    h3(r0, hmx0, hmn0, xv0);
    for (int r = r0; r < r0 + 64; ++r) {
      if (r + 1 < H_) h3(r + 1, hmxp, hmnp, xvp);
      else { hmxp = -FLT_MAX; hmnp = FLT_MAX; xvp = 0.f; }
      float mx = fmaxf(hmxm, fmaxf(hmx0, hmxp));
      float mn = fminf(hmnm, fminf(hmn0, hmnp));
      R0 += fmaxf(mx - xv0, 0.f);
      R1 += fmaxf(xv0 - mn, 0.f);
      hmxm = hmx0; hmnm = hmn0; hmx0 = hmxp; hmn0 = hmnp; xv0 = xvp;
    }
  }
  const float T0 = block_reduce_1024(R0, red, tid);
  const float T1 = block_reduce_1024(R1, red, tid);
  const float s0 = wv * T0 + EPSP;       // s = sum(l) + EPS_P, l = w * r
  const float s1 = wv * T1 + EPSP;

  // ---- pass 2: E = -sum p*log(p+eps), p = (w*r)/s ----
  float E0 = 0.f, E1 = 0.f;
  {
    float hmxm, hmnm, hmx0, hmn0, hmxp, hmnp, xv0, xvp, dummy;
    if (r0 > 0) h3(r0 - 1, hmxm, hmnm, dummy); else { hmxm = -FLT_MAX; hmnm = FLT_MAX; }
    h3(r0, hmx0, hmn0, xv0);
    const float inv_s0 = wv / s0, inv_s1 = wv / s1;
    for (int r = r0; r < r0 + 64; ++r) {
      if (r + 1 < H_) h3(r + 1, hmxp, hmnp, xvp);
      else { hmxp = -FLT_MAX; hmnp = FLT_MAX; xvp = 0.f; }
      float mx = fmaxf(hmxm, fmaxf(hmx0, hmxp));
      float mn = fminf(hmnm, fminf(hmn0, hmnp));
      float p0 = fmaxf(mx - xv0, 0.f) * inv_s0;
      float p1 = fmaxf(xv0 - mn, 0.f) * inv_s1;
      E0 -= p0 * __logf(p0 + EPSP);
      E1 -= p1 * __logf(p1 + EPSP);
      hmxm = hmx0; hmnm = hmn0; hmx0 = hmxp; hmn0 = hmnp; xv0 = xvp;
    }
  }
  const float F0 = block_reduce_1024(E0, red, tid);
  const float F1 = block_reduce_1024(E1, red, tid);
  if (tid == 0) {
    feats[slice * 2 + 0] = F0;
    feats[slice * 2 + 1] = F1;
  }
}

// ---------------------------------------------------------------------------
// Kernel 4: f = sum_c feats*w_norm; logits = relu(f@cls1.T+b)@cls2.T+b (WMMA)
// ---------------------------------------------------------------------------
__global__ void k_cls(const float* __restrict__ feats, const float* __restrict__ w_ws,
                      const float* __restrict__ cls1_w, const float* __restrict__ cls1_b,
                      const float* __restrict__ cls2_w, const float* __restrict__ cls2_b,
                      float* __restrict__ out_logits, float* __restrict__ out_f) {
  __shared__ float fA[B_][2];
  __shared__ float h2[B_][MLP_];
  const int lane = threadIdx.x;          // one wave

  {
    float sw = 0.f;
    for (int c = 0; c < C_; ++c) sw += w_ws[lane * C_ + c];
    const float den = sw + EPSW;
    float f0 = 0.f, f1 = 0.f;
    for (int c = 0; c < C_; ++c) {
      float wn = w_ws[lane * C_ + c] / den;
      f0 += feats[(lane * C_ + c) * 2 + 0] * wn;
      f1 += feats[(lane * C_ + c) * 2 + 1] * wn;
    }
    fA[lane][0] = f0; fA[lane][1] = f1;
    out_f[lane * 2 + 0] = f0; out_f[lane * 2 + 1] = f1;
  }
  __syncthreads();

  const int m15 = lane & 15;
  const int hsel = (lane < 16) ? 0 : 8;

  // GEMM3: [32,2(pad32)] x [2,128] -> [32,128], relu
  for (int mt = 0; mt < 2; ++mt)
    for (int nt = 0; nt < 8; ++nt) {
      v8f acc = {};
      v16h a, b;
      for (int e = 0; e < 16; ++e) {
        int k = kmap(lane, e);
        a[e] = (k < 2) ? (_Float16)fA[mt * 16 + m15][k] : (_Float16)0.f;
        b[e] = (k < 2) ? (_Float16)cls1_w[(nt * 16 + m15) * 2 + k] : (_Float16)0.f;
      }
      acc = __builtin_amdgcn_wmma_f32_16x16x32_f16(false, a, false, b,
                                                   (short)0, acc, false, false);
      for (int v = 0; v < 8; ++v) {
        int m = mt * 16 + hsel + v, n = nt * 16 + m15;
        float t = acc[v] + cls1_b[n];
        h2[m][n] = t > 0.f ? t : 0.f;
      }
    }
  __syncthreads();

  // GEMM4: [32,128] x [128,5(pad16)] -> [32,5]
  for (int mt = 0; mt < 2; ++mt) {
    v8f acc = {};
    for (int kc = 0; kc < 4; ++kc) {
      v16h a, b;
      for (int e = 0; e < 16; ++e) {
        int k = kc * 32 + kmap(lane, e);
        a[e] = (_Float16)h2[mt * 16 + m15][k];
        b[e] = (m15 < NC_) ? (_Float16)cls2_w[m15 * MLP_ + k] : (_Float16)0.f;
      }
      acc = __builtin_amdgcn_wmma_f32_16x16x32_f16(false, a, false, b,
                                                   (short)0, acc, false, false);
    }
    for (int v = 0; v < 8; ++v) {
      int m = mt * 16 + hsel + v, n = m15;
      if (n < NC_) out_logits[m * NC_ + n] = acc[v] + cls2_b[n];
    }
  }
}

// ---------------------------------------------------------------------------
extern "C" void kernel_launch(void* const* d_in, const int* in_sizes, int n_in,
                              void* d_out, int out_size, void* d_ws, size_t ws_size,
                              hipStream_t stream) {
  (void)in_sizes; (void)n_in; (void)out_size; (void)ws_size;
  const float* x      = (const float*)d_in[0];
  const float* fc1_w  = (const float*)d_in[1];
  const float* fc1_b  = (const float*)d_in[2];
  const float* fc2_w  = (const float*)d_in[3];
  const float* fc2_b  = (const float*)d_in[4];
  const float* cls1_w = (const float*)d_in[5];
  const float* cls1_b = (const float*)d_in[6];
  const float* cls2_w = (const float*)d_in[7];
  const float* cls2_b = (const float*)d_in[8];

  float* out    = (float*)d_out;                 // [logits 160 | w 2048 | f 64]
  float* logits = out;
  float* w_out  = out + B_ * NC_;
  float* f_out  = out + B_ * NC_ + B_ * C_;

  float* ws    = (float*)d_ws;
  float* z_ws  = ws;                             // 2048 floats
  float* w_ws  = ws + 2048;                      // 2048 floats
  float* feats = ws + 4096;                      // 4096 floats

  k_mean<<<B_ * C_, 256, 0, stream>>>(x, z_ws);
  k_attn<<<1, 32, 0, stream>>>(z_ws, fc1_w, fc1_b, fc2_w, fc2_b, w_ws, w_out);

  const size_t shbytes = (size_t)HW_ * sizeof(float);   // 256 KB dynamic LDS
  (void)hipFuncSetAttribute((const void*)k_topo,
                            hipFuncAttributeMaxDynamicSharedMemorySize,
                            (int)shbytes);
  k_topo<<<B_ * C_, 1024, shbytes, stream>>>(x, w_ws, feats);

  k_cls<<<1, 32, 0, stream>>>(feats, w_ws, cls1_w, cls1_b, cls2_w, cls2_b,
                              logits, f_out);
}
